// GINEConvEncoder_12489764896953
// MI455X (gfx1250) — compile-verified, matrix operand order checked by reading, not run on hardware
//
#include <hip/hip_runtime.h>

// ---------------------------------------------------------------------------
// Types for CDNA5 WMMA
// ---------------------------------------------------------------------------
typedef __attribute__((ext_vector_type(16))) __bf16 v16bf;
typedef __attribute__((ext_vector_type(8)))  float  v8f;
typedef __attribute__((ext_vector_type(4)))  unsigned int u32x4;

union BFrag { v16bf v; u32x4 u[2]; };

#define HDIM 256
#define EDGES 131072
#define NGRAPH 64

__device__ __forceinline__ unsigned short f2bf(float f) {
    unsigned int u = __float_as_uint(f);
    unsigned int r = u + 0x7FFFu + ((u >> 16) & 1u);   // round-to-nearest-even
    return (unsigned short)(r >> 16);
}

// ---------------------------------------------------------------------------
// Utility fills / converts
// ---------------------------------------------------------------------------
__global__ void fill_f32(float* p, float v, int n) {
    int i = blockIdx.x * 256 + threadIdx.x;
    if (i < n) p[i] = v;
}
__global__ void fill_i32(int* p, int v, int n) {
    int i = blockIdx.x * 256 + threadIdx.x;
    if (i < n) p[i] = v;
}
__global__ void add_cvt_bf16(const float* __restrict__ a, const float* __restrict__ b,
                             unsigned short* __restrict__ o, int n) {
    int i = blockIdx.x * 256 + threadIdx.x;
    if (i < n) o[i] = f2bf(a[i] + b[i]);
}
__global__ void cvt_bf16(const float* __restrict__ a, unsigned short* __restrict__ o, int n) {
    int i = blockIdx.x * 256 + threadIdx.x;
    if (i < n) o[i] = f2bf(a[i]);
}
// Wt[n*256+k] = bf16(W[k*256+n])  (one-time; keeps GEMM B loads contiguous in K)
__global__ void transpose_cvt(const float* __restrict__ W, unsigned short* __restrict__ Wt) {
    int i = blockIdx.x * 256 + threadIdx.x;   // 65536 total
    int n = i & 255, k = i >> 8;
    Wt[n * HDIM + k] = f2bf(W[k * HDIM + n]);
}
__global__ void init_edges(const int* __restrict__ ei, int* src, int* dst, int* mask) {
    int e = blockIdx.x * 256 + threadIdx.x;
    if (e < EDGES) { src[e] = ei[e]; dst[e] = ei[EDGES + e]; mask[e] = 1; }
}

// ---------------------------------------------------------------------------
// Edge stage: e = ea@We+be ; m = relu(x[src]+e) ; agg[dst] += m (masked)
// ---------------------------------------------------------------------------
__global__ void edge_agg(const float* __restrict__ x, const float* __restrict__ ea,
                         const int* __restrict__ src, const int* __restrict__ dst,
                         const int* __restrict__ mask,
                         const float* __restrict__ We, const float* __restrict__ be,
                         float* __restrict__ agg) {
    int e = blockIdx.x;
    if (!mask[e]) return;
    int h = threadIdx.x;
    int s = src[e], d = dst[e];
    __builtin_prefetch(&x[(size_t)s * HDIM + h], 0, 0);   // global_prefetch_b8
    float a0 = ea[e * 2 + 0], a1 = ea[e * 2 + 1];
    float eh = fmaf(a0, We[h], fmaf(a1, We[HDIM + h], be[h]));
    float v = x[(size_t)s * HDIM + h] + eh;
    v = fmaxf(v, 0.0f);
    atomicAdd(&agg[(size_t)d * HDIM + h], v);
}

// ---------------------------------------------------------------------------
// WMMA GEMM: C[M,256] = act(A_bf16[M,256] @ Wt_bf16^T + bias)
// Block = 256 threads (8 waves); block computes 16 rows x 256 cols.
// Each wave: two 16x16 tiles; K-loop 8 x v_wmma_f32_16x16x32_bf16.
// 16-bit A/B fragment striping per CDNA5 ISA 7.12.2:
//   lanes 0-15 hold K = k0+[0..7] , k0+[16..23]; lanes 16-31 hold K = k0+[8..15], k0+[24..31]
// ---------------------------------------------------------------------------
__global__ void gemm_bf16_h256(const unsigned short* __restrict__ A,
                               const unsigned short* __restrict__ Wt,
                               const float* __restrict__ bias,
                               float* __restrict__ C, int doRelu) {
    const int K = HDIM;
    int tid  = threadIdx.x;
    int wave = tid >> 5;
    int lane = tid & 31;
    int half = lane >> 4;
    int m    = lane & 15;
    int rowBase = blockIdx.x * 16;
    int nBase   = wave * 32;

    const unsigned short* arow = A  + (size_t)(rowBase + m) * K;
    const unsigned short* br0  = Wt + (size_t)(nBase + m) * K;
    const unsigned short* br1  = Wt + (size_t)(nBase + 16 + m) * K;

    v8f c0 = {}; v8f c1 = {};
    #pragma unroll
    for (int k0 = 0; k0 < K; k0 += 32) {
        int kLo = k0 + (half ? 8 : 0);
        int kHi = k0 + (half ? 24 : 16);
        BFrag av, b0, b1;
        av.u[0] = *(const u32x4*)(arow + kLo);
        av.u[1] = *(const u32x4*)(arow + kHi);
        b0.u[0] = *(const u32x4*)(br0 + kLo);
        b0.u[1] = *(const u32x4*)(br0 + kHi);
        b1.u[0] = *(const u32x4*)(br1 + kLo);
        b1.u[1] = *(const u32x4*)(br1 + kHi);
        c0 = __builtin_amdgcn_wmma_f32_16x16x32_bf16(false, av.v, false, b0.v,
                                                     (short)0, c0, false, false);
        c1 = __builtin_amdgcn_wmma_f32_16x16x32_bf16(false, av.v, false, b1.v,
                                                     (short)0, c1, false, false);
    }
    // C/D layout: VGPR r -> (M = r + 8*half, N = nBase + m)
    int col = nBase + m;
    float bb0 = bias[col], bb1 = bias[col + 16];
    #pragma unroll
    for (int r = 0; r < 8; r++) {
        int row = rowBase + r + half * 8;
        float v0 = c0[r] + bb0;
        float v1 = c1[r] + bb1;
        if (doRelu) { v0 = fmaxf(v0, 0.0f); v1 = fmaxf(v1, 0.0f); }
        C[(size_t)row * HDIM + col]      = v0;
        C[(size_t)row * HDIM + col + 16] = v1;
    }
}

// ---------------------------------------------------------------------------
// score[i] = (x[i,:].p) / ||p||   (one node per wave, 8 waves/block)
// ---------------------------------------------------------------------------
__global__ void score_kernel(const float* __restrict__ x, const float* __restrict__ p,
                             float* __restrict__ score, int n) {
    int wave = threadIdx.x >> 5, lane = threadIdx.x & 31;
    int node = blockIdx.x * 8 + wave;
    if (node >= n) return;
    float dot = 0.f, pn = 0.f;
    #pragma unroll
    for (int j = 0; j < 8; j++) {
        float pv = p[lane + 32 * j];
        dot = fmaf(x[(size_t)node * HDIM + lane + 32 * j], pv, dot);
        pn  = fmaf(pv, pv, pn);
    }
    #pragma unroll
    for (int d = 16; d > 0; d >>= 1) {
        dot += __shfl_down(dot, d, 32);
        pn  += __shfl_down(pn,  d, 32);
    }
    if (lane == 0) score[node] = dot * rsqrtf(pn);
}

// ---------------------------------------------------------------------------
// TopK by rank (descending value, index tie-break) — one block per graph.
// ---------------------------------------------------------------------------
__global__ void topk_kernel(const float* __restrict__ score, int* __restrict__ perm,
                            int n_g, int k) {
    __shared__ float s[256];
    int g = blockIdx.x, t = threadIdx.x;
    int base = g * n_g;
    if (t < n_g) s[t] = score[base + t];
    __syncthreads();
    if (t < n_g) {
        float v = s[t];
        int rank = 0;
        for (int j = 0; j < n_g; j++) {
            float u = s[j];
            rank += (u > v) || (u == v && j < t);
        }
        if (rank < k) perm[g * k + rank] = base + t;
    }
}

__global__ void pool_gather(const float* __restrict__ xo, const float* __restrict__ score,
                            const int* __restrict__ perm, float* __restrict__ xn) {
    int i = blockIdx.x, h = threadIdx.x;
    int old = perm[i];
    xn[(size_t)i * HDIM + h] = xo[(size_t)old * HDIM + h] * tanhf(score[old]);
}

__global__ void set_inv(const int* __restrict__ perm, int* __restrict__ inv, int nnew) {
    int i = blockIdx.x * 256 + threadIdx.x;
    if (i < nnew) inv[perm[i]] = i;
}

__global__ void relabel(int* src, int* dst, int* mask, const int* __restrict__ inv) {
    int e = blockIdx.x * 256 + threadIdx.x;
    if (e >= EDGES) return;
    if (!mask[e]) { src[e] = 0; dst[e] = 0; return; }
    int s = inv[src[e]], d = inv[dst[e]];
    int m = (s >= 0) && (d >= 0);
    mask[e] = m;
    src[e] = m ? s : 0;
    dst[e] = m ? d : 0;
}

// ---------------------------------------------------------------------------
// GlobalAttention pooling: out[g,:] += segment_softmax(x@Wg+bg) weighted sum
// One block (256 threads) per graph; k nodes per graph.
// ---------------------------------------------------------------------------
__global__ void attpool(const float* __restrict__ x, const float* __restrict__ Wg,
                        const float* __restrict__ bg, float* __restrict__ out, int k) {
    __shared__ float gate[256];
    __shared__ float red[256];
    int g = blockIdx.x, t = threadIdx.x;
    const float* xg = x + (size_t)g * k * HDIM;
    float gv = -1e30f;
    if (t < k) {
        float acc = bg[0];
        for (int h = 0; h < HDIM; h++) acc = fmaf(xg[(size_t)t * HDIM + h], Wg[h], acc);
        gv = acc;
    }
    gate[t] = gv;
    red[t]  = gv;
    __syncthreads();
    for (int s2 = 128; s2 > 0; s2 >>= 1) {
        if (t < s2) red[t] = fmaxf(red[t], red[t + s2]);
        __syncthreads();
    }
    float gmax = red[0];
    __syncthreads();
    float ev = (t < k) ? __expf(gate[t] - gmax) : 0.0f;
    red[t] = ev;
    __syncthreads();
    for (int s2 = 128; s2 > 0; s2 >>= 1) {
        if (t < s2) red[t] += red[t + s2];
        __syncthreads();
    }
    float esum = red[0];
    __syncthreads();
    gate[t] = ev / esum;
    __syncthreads();
    float acc = 0.0f;
    for (int j = 0; j < k; j++) acc = fmaf(gate[j], xg[(size_t)j * HDIM + t], acc);
    out[g * HDIM + t] += acc;
}

// ---------------------------------------------------------------------------
// Host orchestration
// ---------------------------------------------------------------------------
extern "C" void kernel_launch(void* const* d_in, const int* in_sizes, int n_in,
                              void* d_out, int out_size, void* d_ws, size_t ws_size,
                              hipStream_t stream) {
    const float* x    = (const float*)d_in[0];
    const float* ea   = (const float*)d_in[1];
    const int*   ei   = (const int*)  d_in[2];
    const float* We   = (const float*)d_in[4];
    const float* be   = (const float*)d_in[5];
    const float* W1   = (const float*)d_in[6];
    const float* b1   = (const float*)d_in[7];
    const float* W2   = (const float*)d_in[8];
    const float* b2   = (const float*)d_in[9];
    const float* pvec[3] = { (const float*)d_in[10], (const float*)d_in[11],
                             (const float*)d_in[12] };
    const float* Wg   = (const float*)d_in[13];
    const float* bg   = (const float*)d_in[14];
    float* out = (float*)d_out;

    // ---- workspace carve-up ----
    char* w = (char*)d_ws;
    const size_t N0   = 16384;
    const size_t XBYT = N0 * HDIM * sizeof(float);          // 16 MB
    float* xA  = (float*)w;               w += XBYT;
    float* xB  = (float*)w;               w += XBYT;
    float* agg = (float*)w;               w += XBYT;
    float* h1  = (float*)w;               w += XBYT;
    unsigned short* abf = (unsigned short*)w; w += N0 * HDIM * sizeof(unsigned short);
    unsigned short* wt1 = (unsigned short*)w; w += HDIM * HDIM * sizeof(unsigned short);
    unsigned short* wt2 = (unsigned short*)w; w += HDIM * HDIM * sizeof(unsigned short);
    int* srcw  = (int*)w; w += EDGES * sizeof(int);
    int* dstw  = (int*)w; w += EDGES * sizeof(int);
    int* maskw = (int*)w; w += EDGES * sizeof(int);
    float* score = (float*)w; w += N0 * sizeof(float);
    int* perm  = (int*)w; w += N0 * sizeof(int);
    int* inv   = (int*)w; w += N0 * sizeof(int);

    // ---- one-time setup ----
    fill_f32<<<(out_size + 255) / 256, 256, 0, stream>>>(out, 0.0f, out_size);
    transpose_cvt<<<(HDIM * HDIM) / 256, 256, 0, stream>>>(W1, wt1);
    transpose_cvt<<<(HDIM * HDIM) / 256, 256, 0, stream>>>(W2, wt2);
    hipMemcpyAsync(xA, x, XBYT, hipMemcpyDeviceToDevice, stream);
    init_edges<<<EDGES / 256, 256, 0, stream>>>(ei, srcw, dstw, maskw);

    // layer geometry (RATIO=0.8): n_g 256->205->164, k = ceil(0.8*n_g)
    const int n_arr[3]  = { 16384, 13120, 10496 };
    const int ng_arr[3] = { 256, 205, 164 };
    const int k_arr[3]  = { 205, 164, 132 };

    for (int L = 0; L < 3; L++) {
        const int n  = n_arr[L];
        const int ng = ng_arr[L];
        const int k  = k_arr[L];
        const int nn = NGRAPH * k;
        const int nh = n * HDIM;

        // GINE edge aggregation
        fill_f32<<<(nh + 255) / 256, 256, 0, stream>>>(agg, 0.0f, nh);
        edge_agg<<<EDGES, HDIM, 0, stream>>>(xA, ea, srcw, dstw, maskw, We, be, agg);

        // nn MLP via WMMA: h1 = relu((agg+x)@W1+b1); x = relu(h1@W2+b2)
        add_cvt_bf16<<<(nh + 255) / 256, 256, 0, stream>>>(agg, xA, abf, nh);
        gemm_bf16_h256<<<n / 16, 256, 0, stream>>>(abf, wt1, b1, h1, 1);
        cvt_bf16<<<(nh + 255) / 256, 256, 0, stream>>>(h1, abf, nh);
        gemm_bf16_h256<<<n / 16, 256, 0, stream>>>(abf, wt2, b2, xA, 1);

        // TopK pooling
        score_kernel<<<(n + 7) / 8, 256, 0, stream>>>(xA, pvec[L], score, n);
        topk_kernel<<<NGRAPH, 256, 0, stream>>>(score, perm, ng, k);
        pool_gather<<<nn, HDIM, 0, stream>>>(xA, score, perm, xB);

        // relabel edges for next layer
        fill_i32<<<(n + 255) / 256, 256, 0, stream>>>(inv, -1, n);
        set_inv<<<(nn + 255) / 256, 256, 0, stream>>>(perm, inv, nn);
        relabel<<<EDGES / 256, 256, 0, stream>>>(srcw, dstw, maskw, inv);

        // attention pooling, accumulate into out
        attpool<<<NGRAPH, HDIM, 0, stream>>>(xB, Wg, bg, out, k);

        // swap x buffers
        float* t = xA; xA = xB; xB = t;
    }
}
